// SelfAttention_64252710748698
// MI455X (gfx1250) — matrix operand order repaired and checked by reading, not compiled, not run on hardware
//
#include <hip/hip_runtime.h>
#include <hip/hip_bf16.h>

// Problem constants (from reference): B=16, H=128, W=128, C=256, QK_DIM=32
#define W_DIM 128
#define C_DIM 256
#define QK 32

typedef __attribute__((ext_vector_type(16))) __bf16 v16bf;
typedef __attribute__((ext_vector_type(8)))  float  v8f;
typedef __attribute__((ext_vector_type(4)))  int    v4i;

#if __has_builtin(__builtin_amdgcn_global_load_async_to_lds_b128)
#define HAVE_ASYNC_LDS 1
typedef __attribute__((address_space(1))) v4i* as1_v4i_ptr;
typedef __attribute__((address_space(3))) v4i* as3_v4i_ptr;
#else
#define HAVE_ASYNC_LDS 0
#endif

__device__ __forceinline__ unsigned short f2bf(float f) {
    unsigned int u = __float_as_uint(f);
    u += 0x7FFFu + ((u >> 16) & 1u);        // round-to-nearest-even
    return (unsigned short)(u >> 16);
}

__device__ __forceinline__ uint2 pack4_bf16(float4 f) {
    uint2 r;
    r.x = (unsigned)f2bf(f.x) | ((unsigned)f2bf(f.y) << 16);
    r.y = (unsigned)f2bf(f.z) | ((unsigned)f2bf(f.w) << 16);
    return r;
}

union Frag {
    v16bf v;
    uint4 q[2];
    unsigned short u[16];
};

// A-matrix fragment (16x32 bf16, MxK), row-major [m][k] source, ld in halves.
// ISA layout: lane&15 = row; a[0..7] = K (k0+8*half .. +7); a[8..15] = K (k0+16+8*half .. +7)
__device__ __forceinline__ v16bf load_a(const unsigned short* p, int ld, int m0, int k0, int lane) {
    int h = lane >> 4;
    const unsigned short* base = p + (m0 + (lane & 15)) * ld + k0 + 8 * h;
    Frag f;
    f.q[0] = *(const uint4*)(base);
    f.q[1] = *(const uint4*)(base + 16);
    return f.v;
}

// B-matrix fragment (32x16 bf16, KxN) from an n-major [n][k] source, ld in halves.
// ISA layout: lane&15 = column n; b[0..7] = K (k0+8*half .. +7); b[8..15] = K (k0+16+8*half .. +7)
__device__ __forceinline__ v16bf load_b(const unsigned short* p, int ld, int n0, int k0, int lane) {
    const unsigned short* base = p + (n0 + (lane & 15)) * ld + k0 + 8 * (lane >> 4);
    Frag f;
    f.q[0] = *(const uint4*)(base);
    f.q[1] = *(const uint4*)(base + 16);
    return f.v;
}

#define WMMA_BF16(a, b, c) \
    __builtin_amdgcn_wmma_f32_16x16x32_bf16(false, (a), false, (b), (short)0, (c), false, false)

#if HAVE_ASYNC_LDS
__device__ __forceinline__ void wait_asynccnt_4() {
#if __has_builtin(__builtin_amdgcn_s_wait_asynccnt)
    __builtin_amdgcn_s_wait_asynccnt(4);
#else
    asm volatile("s_wait_asynccnt 0x4" ::: "memory");
#endif
}
__device__ __forceinline__ void wait_asynccnt_0() {
#if __has_builtin(__builtin_amdgcn_s_wait_asynccnt)
    __builtin_amdgcn_s_wait_asynccnt(0);
#else
    asm volatile("s_wait_asynccnt 0x0" ::: "memory");
#endif
}

// Stream one 16-row (16 KB) chunk of X into an LDS staging buffer.
// 256 threads x 16B x 4 issues = 16 KB; each wave's ASYNCcnt rises by 4.
__device__ __forceinline__ void issue_chunk(const float* xg, float* buf, int cs, int tid) {
    const float* gp = xg + cs * 4096 + tid * 4;
    float* lp = buf + tid * 4;
    #pragma unroll
    for (int j = 0; j < 4; ++j) {
        __builtin_amdgcn_global_load_async_to_lds_b128(
            (as1_v4i_ptr)(gp + j * 1024),
            (as3_v4i_ptr)(lp + j * 1024),
            0, 0);
    }
}
#endif

// ---------------------------------------------------------------------------
// Pre-pass: convert weights fp32 -> bf16, transposed to [n][k] so WMMA
// B-fragments are two contiguous 8-half runs per lane. Wq,Wk: [256,32] -> [32][256].
// Wv: [256,256] -> [256][256] transposed. All land in d_ws (L2-resident, 160 KB).
// ---------------------------------------------------------------------------
__global__ void prep_weights(const float* __restrict__ Wq, const float* __restrict__ Wk,
                             const float* __restrict__ Wv,
                             unsigned short* __restrict__ wqT, unsigned short* __restrict__ wkT,
                             unsigned short* __restrict__ wvT) {
    int i = blockIdx.x * blockDim.x + threadIdx.x;
    if (i < QK * C_DIM) {
        int n = i / C_DIM, k = i % C_DIM;
        wqT[n * C_DIM + k] = f2bf(Wq[k * QK + n]);
        wkT[n * C_DIM + k] = f2bf(Wk[k * QK + n]);
    }
    if (i < C_DIM * C_DIM) {
        int n = i / C_DIM, k = i % C_DIM;
        wvT[n * C_DIM + k] = f2bf(Wv[k * C_DIM + n]);
    }
}

// ---------------------------------------------------------------------------
// Fused attention: one workgroup (256 threads = 8 wave32) per (b,h).
// LDS map (192 KB total, lifetime-aliased):
//   [0      , 65536 ) Xb    bf16[128][256]  (X staged)       -- aliased later by:
//   [0      , 65536 ) Sf    f32 [128][128]  (scores, after X dead)
//   [65536  , 73728 ) Qb    bf16[128][32]
//   [73728  , 81920 ) Kb    bf16[128][32]   (row-major == K^T B-fragment layout)
//   [65536  , 98304 ) Pb    bf16[128][128]  (probs; aliases Qb/Kb after S done)
//   [98304  , 163840) Vt    bf16[256][128]  (V transposed -> contiguous B-frags)
//   [163840 , 196608) stage f32 double-buffer for async X streaming;
//                     reused as softmax reduction scratch afterwards.
// ---------------------------------------------------------------------------
__global__ void __launch_bounds__(256)
attn_fused_kernel(const float* __restrict__ x,
                  const float* __restrict__ bq, const float* __restrict__ bk,
                  const float* __restrict__ bv,
                  const unsigned short* __restrict__ wqT,
                  const unsigned short* __restrict__ wkT,
                  const unsigned short* __restrict__ wvT,
                  float* __restrict__ out) {
    __shared__ __align__(16) char smem[196608];
    unsigned short* Xb = (unsigned short*)smem;               // [128][256]
    float*          Sf = (float*)smem;                        // [128][128] (aliases Xb)
    unsigned short* Qb = (unsigned short*)(smem + 65536);     // [128][32]
    unsigned short* Kb = (unsigned short*)(smem + 73728);     // [128][32]
    unsigned short* Pb = (unsigned short*)(smem + 65536);     // [128][128] (aliases Qb/Kb)
    unsigned short* Vt = (unsigned short*)(smem + 98304);     // [256][128]
    float*          stage = (float*)(smem + 163840);          // 2 x 4096 floats

    const int tid  = threadIdx.x;
    const int wv   = tid >> 5;
    const int lane = tid & 31;
    const int h    = lane >> 4;
    const int nl   = lane & 15;

    const float* xg = x + (size_t)blockIdx.x * (W_DIM * C_DIM);

    // ---- Phase 0: stage X[b,h] (fp32 HBM) as bf16 in LDS --------------------
#if HAVE_ASYNC_LDS
    // Double-buffered async pipeline: global->LDS B128 async copies of the
    // next 16-row chunk overlap with bf16 conversion of the current chunk.
    issue_chunk(xg, stage, 0, tid);
    for (int cs = 0; cs < 8; ++cs) {
        if (cs + 1 < 8) {
            issue_chunk(xg, stage + ((cs + 1) & 1) * 4096, cs + 1, tid);
            wait_asynccnt_4();              // chunk cs landed; cs+1 in flight
        } else {
            wait_asynccnt_0();
        }
        __syncthreads();
        const float* buf = stage + (cs & 1) * 4096;
        const int base = tid * 16;          // 16 consecutive floats per thread
        #pragma unroll
        for (int j = 0; j < 4; ++j) {
            float4 f = *(const float4*)(buf + base + j * 4);
            int e = cs * 4096 + base + j * 4;
            *(uint2*)(Xb + (e >> 8) * C_DIM + (e & 255)) = pack4_bf16(f);
        }
        __syncthreads();                    // buffer reusable next iteration
    }
#else
    for (int i = tid * 4; i < W_DIM * C_DIM; i += 1024) {
        float4 f = *(const float4*)(xg + i);
        *(uint2*)(Xb + (i >> 8) * C_DIM + (i & 255)) = pack4_bf16(f);
    }
    __syncthreads();
#endif

    // ---- Phase 1: Q = X*Wq + bq, K = X*Wk + bk  (32 tiles of 16x16) ---------
    for (int t = wv * 4; t < wv * 4 + 4; ++t) {
        const bool isQ = (t < 16);
        const int  tt  = isQ ? t : t - 16;
        const int  mi  = tt >> 1, ni = tt & 1;
        const unsigned short* wT = isQ ? wqT : wkT;
        v8f acc = {};
        for (int k0 = 0; k0 < C_DIM; k0 += 32) {
            v16bf a = load_a(Xb, C_DIM, mi * 16, k0, lane);
            v16bf b = load_b(wT, C_DIM, ni * 16, k0, lane);
            acc = WMMA_BF16(a, b, acc);
        }
        const int   n    = ni * 16 + nl;
        const float bias = (isQ ? bq : bk)[n];
        unsigned short* dst = isQ ? Qb : Kb;
        #pragma unroll
        for (int r = 0; r < 8; ++r) {
            dst[(mi * 16 + r + 8 * h) * QK + n] = f2bf(acc[r] + bias);
        }
    }
    __syncthreads();

    // ---- Phase 2: V = X*Wv + bv, stored transposed Vt[c][w] -----------------
    for (int t = wv; t < 128; t += 8) {
        const int mi = t >> 4, ni = t & 15;
        v8f acc = {};
        for (int k0 = 0; k0 < C_DIM; k0 += 32) {
            v16bf a = load_a(Xb, C_DIM, mi * 16, k0, lane);
            v16bf b = load_b(wvT, C_DIM, ni * 16, k0, lane);
            acc = WMMA_BF16(a, b, acc);
        }
        const int   c    = ni * 16 + nl;
        const float bias = bv[c];
        union { uint4 q; unsigned short u[8]; } pk;
        #pragma unroll
        for (int r = 0; r < 8; ++r) pk.u[r] = f2bf(acc[r] + bias);  // r -> consecutive w
        *(uint4*)(Vt + c * W_DIM + mi * 16 + 8 * h) = pk.q;         // 16B LDS store
    }
    __syncthreads();   // X dead; Sf may now alias its storage

    // ---- Phase 3: S = Q * K^T  (64 tiles, K-dim = 32 -> single WMMA each) ---
    // Kb row-major [w][d] IS K^T's B-fragment layout (lane=col w, contiguous d).
    for (int t = wv; t < 64; t += 8) {
        const int mi = t >> 3, ni = t & 7;
        v16bf a = load_a(Qb, QK, mi * 16, 0, lane);
        v16bf b = load_b(Kb, QK, ni * 16, 0, lane);
        v8f acc = {};
        acc = WMMA_BF16(a, b, acc);
        const int n = ni * 16 + nl;
        #pragma unroll
        for (int r = 0; r < 8; ++r) {
            Sf[(mi * 16 + r + 8 * h) * W_DIM + n] = acc[r];
        }
    }
    __syncthreads();   // Q/K dead; Pb may now alias their storage

    // ---- Phase 4: row softmax, 2 threads per row (64 elems each) ------------
    {
        float* red = stage;                 // staging buffer is dead -> scratch
        const int r = tid & 127, s = tid >> 7;
        float* row = Sf + r * W_DIM + s * 64;

        float mx = -3.402823466e38f;
        #pragma unroll
        for (int j = 0; j < 16; ++j) {
            float4 f = *(const float4*)(row + j * 4);
            mx = fmaxf(mx, fmaxf(fmaxf(f.x, f.y), fmaxf(f.z, f.w)));
        }
        red[tid] = mx;
        __syncthreads();
        mx = fmaxf(red[r], red[r + 128]);

        float sum = 0.0f;
        #pragma unroll
        for (int j = 0; j < 64; ++j) {
            float e = __expf(row[j] - mx);
            row[j] = e;
            sum += e;
        }
        __syncthreads();                    // everyone consumed max partials
        red[tid] = sum;
        __syncthreads();
        const float inv = 1.0f / (red[r] + red[r + 128]);

        #pragma unroll
        for (int j = 0; j < 64; j += 4) {
            float4 f = *(const float4*)(row + j);
            f.x *= inv; f.y *= inv; f.z *= inv; f.w *= inv;
            *(uint2*)(Pb + r * W_DIM + s * 64 + j) = pack4_bf16(f);
        }
    }
    __syncthreads();

    // ---- Phase 5: out = P * V  (128 tiles, K-loop over W=128) ---------------
    float* og = out + (size_t)blockIdx.x * (W_DIM * C_DIM);
    for (int t = wv; t < 128; t += 8) {
        const int mi = t >> 4, ni = t & 15;
        v8f acc = {};
        for (int k0 = 0; k0 < W_DIM; k0 += 32) {
            v16bf a = load_a(Pb, W_DIM, mi * 16, k0, lane);
            v16bf b = load_b(Vt, W_DIM, ni * 16, k0, lane);
            acc = WMMA_BF16(a, b, acc);
        }
        const int c = ni * 16 + nl;
        #pragma unroll
        for (int r = 0; r < 8; ++r) {
            og[(size_t)(mi * 16 + r + 8 * h) * C_DIM + c] = acc[r];
        }
    }
}

extern "C" void kernel_launch(void* const* d_in, const int* in_sizes, int n_in,
                              void* d_out, int out_size, void* d_ws, size_t ws_size,
                              hipStream_t stream) {
    (void)in_sizes; (void)n_in; (void)out_size; (void)ws_size;
    const float* x  = (const float*)d_in[0];
    const float* Wq = (const float*)d_in[1];
    const float* bq = (const float*)d_in[2];
    const float* Wk = (const float*)d_in[3];
    const float* bk = (const float*)d_in[4];
    const float* Wv = (const float*)d_in[5];
    const float* bv = (const float*)d_in[6];
    float* out = (float*)d_out;

    unsigned short* wqT = (unsigned short*)d_ws;          //  32*256 bf16
    unsigned short* wkT = wqT + QK * C_DIM;               //  32*256 bf16
    unsigned short* wvT = wkT + QK * C_DIM;               // 256*256 bf16

    prep_weights<<<(C_DIM * C_DIM + 255) / 256, 256, 0, stream>>>(Wq, Wk, Wv, wqT, wkT, wvT);

    const int num_bh = 16 * 128;   // B*H workgroups, one per attention row
    attn_fused_kernel<<<num_bh, 256, 0, stream>>>(x, bq, bk, bv, wqT, wkT, wvT, out);
}
